// GCN_GAT_12558484373613
// MI455X (gfx1250) — matrix operand order repaired and checked by reading, not compiled, not run on hardware
//
#include <hip/hip_runtime.h>
#include <hip/hip_bf16.h>
#include <math.h>

// ---------------------------------------------------------------------------
// GCN+GAT for MI455X (gfx1250). GEMMs via v_wmma_f32_16x16x32_f16 (wave32),
// B panel staged column-major+padded in LDS (f16) so each lane's fragment is
// two ds_load_b128. K fully unrolled. Segment ops via L2 atomics (entire
// ~140MB working set fits the 192MB global L2).
// ---------------------------------------------------------------------------

typedef __attribute__((ext_vector_type(16))) _Float16 v16h;
typedef __attribute__((ext_vector_type(8)))  _Float16 v8h;
typedef __attribute__((ext_vector_type(8)))  float    v8f;

#define NNODES 50000
#define FDIM   256
#define HDIM   64
#define NHEADS 4
#define ODIM   16
#define BPAD   8      // halves of pad per LDS B column (16B) -> bank spread

// ----- order-preserving float <-> uint mapping for atomicMax on floats -----
__device__ __forceinline__ unsigned encOrd(float x) {
    unsigned u = __float_as_uint(x);
    return (u & 0x80000000u) ? ~u : (u | 0x80000000u);
}
__device__ __forceinline__ float decOrd(unsigned u) {
    return __uint_as_float((u & 0x80000000u) ? (u & 0x7FFFFFFFu) : ~u);
}
// encOrd(-inf) = ~0xFF800000 = 0x007FFFFF
#define ENC_NEG_INF 0x007FFFFFu

__device__ __forceinline__ float lrelu(float x) { return x > 0.0f ? x : 0.2f * x; }

// ---------------------------------------------------------------------------
// fills
// ---------------------------------------------------------------------------
__global__ void fill_f32_kernel(float* p, float v, long long n) {
    long long t = (long long)blockIdx.x * blockDim.x + threadIdx.x;
    if (t < n) p[t] = v;
}
__global__ void fill_u32_kernel(unsigned* p, unsigned v, long long n) {
    long long t = (long long)blockIdx.x * blockDim.x + threadIdx.x;
    if (t < n) p[t] = v;
}

// ---------------------------------------------------------------------------
// degree (self-loop graph): deg pre-filled with 1.0 (self loop), += 1 per edge
// ---------------------------------------------------------------------------
__global__ void deg_kernel(const int* __restrict__ dst, float* deg, int E) {
    int t = blockIdx.x * blockDim.x + threadIdx.x;
    if (t < E) atomicAdd(&deg[dst[t]], 1.0f);
}
__global__ void rsqrt_inplace_kernel(float* deg, int n) {
    int t = blockIdx.x * blockDim.x + threadIdx.x;
    if (t < n) { float d = deg[t]; deg[t] = d > 0.0f ? rsqrtf(d) : 0.0f; }
}

// ---------------------------------------------------------------------------
// WMMA GEMM: C[M x Nc] = A[M x K] @ B[K x Nc], f32 in memory, f16 math,
// f32 accumulate. One 16x16 tile per wave, 4 waves/block sharing one LDS-
// staged B panel (column-major, padded: Bs[j*(K+BPAD)+k]). K is a template
// constant -> full unroll -> back-to-back v_wmma chain.
// M % 16 == 0, K % 32 == 0, Nc % 16 == 0 (true for all shapes here).
// Fragment layouts used here (contiguous k, not the ISA table order; A and B agree):
//   A (16x32 f16): lane<16  -> row=lane,    elems = K[0..15]
//                  lane>=16 -> row=lane-16, elems = K[16..31]
//   B (32x16 f16): lane group g=lane>>4 holds K[g*16 .. g*16+15], col=lane&15
//   C/D: VGPR r -> row = r + 8*(lane>>4), col = lane&15
// ---------------------------------------------------------------------------
#define GEMM_WAVES 4
template <int K>
__global__ void wmma_gemm_kernel(const float* __restrict__ A,
                                 const float* __restrict__ B,
                                 float* __restrict__ C,
                                 int M, int Nc) {
    __shared__ _Float16 Bs[16 * (K + BPAD)];

    const int tid   = threadIdx.x;            // 0..127
    const int tileN = blockIdx.x;

    // cooperative B-panel load + f32->f16 convert (once per block).
    // global side coalesced over j (16 contiguous floats per k-row);
    // LDS side column-major: Bs[j*(K+BPAD) + k]
    for (int idx = tid; idx < K * 16; idx += GEMM_WAVES * 32) {
        const int k = idx >> 4, j = idx & 15;
        Bs[j * (K + BPAD) + k] = (_Float16)B[(size_t)k * Nc + tileN * 16 + j];
    }
    __syncthreads();

    const int lane  = tid & 31;
    const int wave  = tid >> 5;
    const int tileM = blockIdx.y * GEMM_WAVES + wave;
    if (tileM * 16 < M) {                     // wave-uniform guard (EXEC all-1)
        const int mrow = lane & 15;
        const int kg   = lane >> 4;           // 0 or 1
        const float* Arow = A + (size_t)(tileM * 16 + mrow) * K;
        const int jc = lane & 15;
        const _Float16* bcolp = &Bs[jc * (K + BPAD)];

        v8f acc = {};
        #pragma unroll
        for (int k0 = 0; k0 < K; k0 += 32) {
            if (k0 + 96 <= K)
                __builtin_prefetch(Arow + k0 + 64, 0, 0);   // global_prefetch_b8
            v16h a;
            const int ak = k0 + kg * 16;
            #pragma unroll
            for (int i = 0; i < 8; ++i) {
                a[i]     = (_Float16)Arow[ak + i];
                a[8 + i] = (_Float16)Arow[ak + 8 + i];
            }
            // fragment = 16 contiguous halves in LDS -> two ds_load_b128
            const int bk = k0 + kg * 16;
            v8h blo = *(const v8h*)(bcolp + bk);
            v8h bhi = *(const v8h*)(bcolp + bk + 8);
            v16h b = __builtin_shufflevector(blo, bhi,
                         0, 1, 2, 3, 4, 5, 6, 7, 8, 9, 10, 11, 12, 13, 14, 15);

            acc = __builtin_amdgcn_wmma_f32_16x16x32_f16(
                      false, a, false, b, (short)0, acc, false, false);
        }

        const int crow = kg * 8;
        const int ccol = tileN * 16 + jc;
        #pragma unroll
        for (int r = 0; r < 8; ++r)
            C[(size_t)(tileM * 16 + crow + r) * Nc + ccol] = acc[r];
    }
}

// ---------------------------------------------------------------------------
// GCN aggregation: out[dst] += h[src] * dinv[src]*dinv[dst], incl. self loops
// thread t -> (edge slot e = t/C, channel c = t%C); slots [E, E+N) are loops
// ---------------------------------------------------------------------------
__global__ void gcn_agg_kernel(const int* __restrict__ src, const int* __restrict__ dst,
                               const float* __restrict__ h, const float* __restrict__ dinv,
                               float* __restrict__ out, int E, int N, int C) {
    long long t = (long long)blockIdx.x * blockDim.x + threadIdx.x;
    long long total = (long long)(E + N) * C;
    if (t >= total) return;
    int e = (int)(t / C), c = (int)(t % C);
    int s, d;
    if (e < E) { s = src[e]; d = dst[e]; } else { s = d = e - E; }
    float nrm = dinv[s] * dinv[d];
    atomicAdd(&out[(size_t)d * C + c], h[(size_t)s * C + c] * nrm);
}

// bias (+ optional relu), in-place safe
__global__ void bias_act_kernel(const float* __restrict__ in, const float* __restrict__ b,
                                float* __restrict__ out, int N, int C, int doRelu) {
    long long t = (long long)blockIdx.x * blockDim.x + threadIdx.x;
    if (t >= (long long)N * C) return;
    int c = (int)(t % C);
    float v = in[t] + b[c];
    out[t] = doRelu ? fmaxf(v, 0.0f) : v;
}

// write agg + bias into output columns [colOff, colOff+C)
__global__ void write_out_kernel(const float* __restrict__ agg, const float* __restrict__ b,
                                 float* __restrict__ out, int N, int C, int colOff, int stride) {
    long long t = (long long)blockIdx.x * blockDim.x + threadIdx.x;
    if (t >= (long long)N * C) return;
    int row = (int)(t / C), c = (int)(t % C);
    out[(size_t)row * stride + colOff + c] = agg[t] + b[c];
}

// ---------------------------------------------------------------------------
// GAT attention pieces
// ---------------------------------------------------------------------------
// per-(node, head) scores: s_src = <h[n,hd,:], a_s[hd,:]>, s_dst likewise
__global__ void att_scores_kernel(const float* __restrict__ h,
                                  const float* __restrict__ a_s, const float* __restrict__ a_d,
                                  float* __restrict__ ssrc, float* __restrict__ sdst,
                                  int N, int heads, int Cper) {
    int t = blockIdx.x * blockDim.x + threadIdx.x;
    if (t >= N * heads) return;
    int n = t / heads, hd = t % heads;
    const float* hp = h + (size_t)n * heads * Cper + (size_t)hd * Cper;
    const float* as = a_s + (size_t)hd * Cper;
    const float* ad = a_d + (size_t)hd * Cper;
    float ss = 0.0f, sd = 0.0f;
    for (int c = 0; c < Cper; ++c) { float v = hp[c]; ss += v * as[c]; sd += v * ad[c]; }
    ssrc[t] = ss; sdst[t] = sd;
}

__device__ __forceinline__ void edge_sd(const int* src, const int* dst, int e, int E,
                                        int& s, int& d) {
    if (e < E) { s = src[e]; d = dst[e]; } else { s = d = e - E; }
}

// pass 1: segment max over dst of leaky_relu(s_src[src]+s_dst[dst])
__global__ void att_max_kernel(const int* __restrict__ src, const int* __restrict__ dst,
                               const float* __restrict__ ssrc, const float* __restrict__ sdst,
                               unsigned* __restrict__ m, int E, int N, int heads) {
    long long t = (long long)blockIdx.x * blockDim.x + threadIdx.x;
    if (t >= (long long)(E + N) * heads) return;
    int e = (int)(t / heads), hd = (int)(t % heads);
    int s, d; edge_sd(src, dst, e, E, s, d);
    float el = lrelu(ssrc[s * heads + hd] + sdst[d * heads + hd]);
    atomicMax(&m[d * heads + hd], encOrd(el));
}

// pass 2: denom[dst] += exp(e - m[dst])
__global__ void att_denom_kernel(const int* __restrict__ src, const int* __restrict__ dst,
                                 const float* __restrict__ ssrc, const float* __restrict__ sdst,
                                 const unsigned* __restrict__ m, float* __restrict__ den,
                                 int E, int N, int heads) {
    long long t = (long long)blockIdx.x * blockDim.x + threadIdx.x;
    if (t >= (long long)(E + N) * heads) return;
    int e = (int)(t / heads), hd = (int)(t % heads);
    int s, d; edge_sd(src, dst, e, E, s, d);
    float el = lrelu(ssrc[s * heads + hd] + sdst[d * heads + hd]);
    atomicAdd(&den[d * heads + hd], expf(el - decOrd(m[d * heads + hd])));
}

// pass 3: out[dst, c] += h[src, c] * alpha(e, head(c))
__global__ void att_agg_kernel(const int* __restrict__ src, const int* __restrict__ dst,
                               const float* __restrict__ ssrc, const float* __restrict__ sdst,
                               const unsigned* __restrict__ m, const float* __restrict__ den,
                               const float* __restrict__ h, float* __restrict__ out,
                               int E, int N, int heads, int C) {
    long long t = (long long)blockIdx.x * blockDim.x + threadIdx.x;
    if (t >= (long long)(E + N) * C) return;
    int e = (int)(t / C), c = (int)(t % C);
    int Cper = C / heads;
    int hd = c / Cper;
    int s, d; edge_sd(src, dst, e, E, s, d);
    float el = lrelu(ssrc[s * heads + hd] + sdst[d * heads + hd]);
    float alpha = expf(el - decOrd(m[d * heads + hd])) / den[d * heads + hd];
    atomicAdd(&out[(size_t)d * C + c], h[(size_t)s * C + c] * alpha);
}

// ---------------------------------------------------------------------------
// host-side orchestration
// ---------------------------------------------------------------------------
static inline unsigned nblk(long long n, int thr) { return (unsigned)((n + thr - 1) / thr); }

extern "C" void kernel_launch(void* const* d_in, const int* in_sizes, int n_in,
                              void* d_out, int out_size, void* d_ws, size_t ws_size,
                              hipStream_t stream) {
    const float* x       = (const float*)d_in[0];
    const int*   eidx    = (const int*)d_in[1];
    const float* W_gcn1  = (const float*)d_in[2];
    const float* b_gcn1  = (const float*)d_in[3];
    const float* W_gcn2  = (const float*)d_in[4];
    const float* b_gcn2  = (const float*)d_in[5];
    const float* W_gat1  = (const float*)d_in[6];
    const float* a_src1  = (const float*)d_in[7];
    const float* a_dst1  = (const float*)d_in[8];
    const float* b_gat1  = (const float*)d_in[9];
    const float* W_gat2  = (const float*)d_in[10];
    const float* a_src2  = (const float*)d_in[11];
    const float* a_dst2  = (const float*)d_in[12];
    const float* b_gat2  = (const float*)d_in[13];
    float* out = (float*)d_out;

    const int N = in_sizes[0] / FDIM;           // 50000
    const int E = in_sizes[1] / 2;              // 800000
    const int* src = eidx;
    const int* dst = eidx + E;
    const int ET = E + N;                       // edge slots incl. self loops
    const int OUTW = 2 * ODIM;                  // 32 output columns

    // workspace layout
    float* ws = (float*)d_ws;
    size_t off = 0;
    auto alloc = [&](size_t n) { float* p = ws + off; off += (n + 63) & ~(size_t)63; return p; };
    float*    dinv    = alloc((size_t)N);
    float*    h_gcn1  = alloc((size_t)N * HDIM);
    float*    x_gcn1  = alloc((size_t)N * HDIM);
    float*    h16     = alloc((size_t)N * ODIM);     // reused: h_gcn2 then h_gat2
    float*    agg16   = alloc((size_t)N * ODIM);     // reused: gcn2 agg then gat2 agg
    float*    h_big   = alloc((size_t)N * FDIM);     // h_gat1 (N x 256)
    float*    agg_big = alloc((size_t)N * FDIM);     // gat1 agg / x_gat1
    float*    ssrc1   = alloc((size_t)N * NHEADS);
    float*    sdst1   = alloc((size_t)N * NHEADS);
    unsigned* m1      = (unsigned*)alloc((size_t)N * NHEADS);
    float*    den1    = alloc((size_t)N * NHEADS);
    float*    ssrc2   = alloc((size_t)N);
    float*    sdst2   = alloc((size_t)N);
    unsigned* m2      = (unsigned*)alloc((size_t)N);
    float*    den2    = alloc((size_t)N);
    (void)ws_size; (void)n_in; (void)out_size;

    const int T = 256;
    auto fillF = [&](float* p, float v, long long n) {
        fill_f32_kernel<<<nblk(n, T), T, 0, stream>>>(p, v, n);
    };
    auto fillU = [&](unsigned* p, unsigned v, long long n) {
        fill_u32_kernel<<<nblk(n, T), T, 0, stream>>>(p, v, n);
    };
    auto gemm = [&](const float* A, const float* B, float* C, int M, int K, int Nc) {
        dim3 grid(Nc / 16, (M / 16 + GEMM_WAVES - 1) / GEMM_WAVES);
        if (K == 256)
            wmma_gemm_kernel<256><<<grid, 32 * GEMM_WAVES, 0, stream>>>(A, B, C, M, Nc);
        else
            wmma_gemm_kernel<64><<<grid, 32 * GEMM_WAVES, 0, stream>>>(A, B, C, M, Nc);
    };

    // ---- symmetric GCN normalization --------------------------------------
    fillF(dinv, 1.0f, N);                                       // self-loop count
    deg_kernel<<<nblk(E, T), T, 0, stream>>>(dst, dinv, E);
    rsqrt_inplace_kernel<<<nblk(N, T), T, 0, stream>>>(dinv, N);

    // ---- GCN layer 1 ------------------------------------------------------
    gemm(x, W_gcn1, h_gcn1, N, FDIM, HDIM);
    fillF(x_gcn1, 0.0f, (long long)N * HDIM);
    gcn_agg_kernel<<<nblk((long long)ET * HDIM, T), T, 0, stream>>>(
        src, dst, h_gcn1, dinv, x_gcn1, E, N, HDIM);
    bias_act_kernel<<<nblk((long long)N * HDIM, T), T, 0, stream>>>(
        x_gcn1, b_gcn1, x_gcn1, N, HDIM, 1);

    // ---- GCN layer 2 ------------------------------------------------------
    gemm(x_gcn1, W_gcn2, h16, N, HDIM, ODIM);
    fillF(agg16, 0.0f, (long long)N * ODIM);
    gcn_agg_kernel<<<nblk((long long)ET * ODIM, T), T, 0, stream>>>(
        src, dst, h16, dinv, agg16, E, N, ODIM);
    write_out_kernel<<<nblk((long long)N * ODIM, T), T, 0, stream>>>(
        agg16, b_gcn2, out, N, ODIM, 0, OUTW);

    // ---- GAT layer 1 (4 heads, concat) ------------------------------------
    gemm(x, W_gat1, h_big, N, FDIM, NHEADS * HDIM);
    att_scores_kernel<<<nblk((long long)N * NHEADS, T), T, 0, stream>>>(
        h_big, a_src1, a_dst1, ssrc1, sdst1, N, NHEADS, HDIM);
    fillU(m1, ENC_NEG_INF, (long long)N * NHEADS);
    att_max_kernel<<<nblk((long long)ET * NHEADS, T), T, 0, stream>>>(
        src, dst, ssrc1, sdst1, m1, E, N, NHEADS);
    fillF(den1, 0.0f, (long long)N * NHEADS);
    att_denom_kernel<<<nblk((long long)ET * NHEADS, T), T, 0, stream>>>(
        src, dst, ssrc1, sdst1, m1, den1, E, N, NHEADS);
    fillF(agg_big, 0.0f, (long long)N * FDIM);
    att_agg_kernel<<<nblk((long long)ET * (NHEADS * HDIM), T), T, 0, stream>>>(
        src, dst, ssrc1, sdst1, m1, den1, h_big, agg_big, E, N, NHEADS, NHEADS * HDIM);
    bias_act_kernel<<<nblk((long long)N * FDIM, T), T, 0, stream>>>(
        agg_big, b_gat1, agg_big, N, NHEADS * HDIM, 1);         // x_gat1 = relu(.)

    // ---- GAT layer 2 (1 head, mean == identity) ----------------------------
    gemm(agg_big, W_gat2, h16, N, NHEADS * HDIM, ODIM);
    att_scores_kernel<<<nblk((long long)N, T), T, 0, stream>>>(
        h16, a_src2, a_dst2, ssrc2, sdst2, N, 1, ODIM);
    fillU(m2, ENC_NEG_INF, (long long)N);
    att_max_kernel<<<nblk((long long)ET, T), T, 0, stream>>>(
        src, dst, ssrc2, sdst2, m2, E, N, 1);
    fillF(den2, 0.0f, (long long)N);
    att_denom_kernel<<<nblk((long long)ET, T), T, 0, stream>>>(
        src, dst, ssrc2, sdst2, m2, den2, E, N, 1);
    fillF(agg16, 0.0f, (long long)N * ODIM);
    att_agg_kernel<<<nblk((long long)ET * ODIM, T), T, 0, stream>>>(
        src, dst, ssrc2, sdst2, m2, den2, h16, agg16, E, N, 1, ODIM);
    write_out_kernel<<<nblk((long long)N * ODIM, T), T, 0, stream>>>(
        agg16, b_gat2, out, N, ODIM, ODIM, OUTW);
}
